// FWGAN500Cont_79096117723370
// MI455X (gfx1250) — compile-verified, matrix operand order checked by reading, not run on hardware
//
#include <hip/hip_runtime.h>
#include <hip/hip_bf16.h>
#include <math.h>

typedef _Float16 half_t;
typedef __attribute__((ext_vector_type(16))) _Float16 v16h;
typedef __attribute__((ext_vector_type(8)))  float    v8f;

#define B_ 32
#define S_ 400
#define T_ 2000
#define PI_ 3.14159265358979323846f

union Frag16 { v16h v; uint4 q[2]; };

__device__ __forceinline__ float sigmoidf_(float x) { return 1.0f / (1.0f + expf(-x)); }

// A fragment (16x32 f16, row-major source). Per ISA: lane L holds row m=L%16,
// k-base = (L/16)*8; VGPR0-3 = k..k+7 (16B), VGPR4-7 = k+16..k+23 (16B).
__device__ __forceinline__ v16h load_a(const half_t* rowp, int koff) {
  Frag16 f;
  f.q[0] = *(const uint4*)(rowp + koff);
  f.q[1] = *(const uint4*)(rowp + koff + 16);
  return f.v;
}

// B fragment from pre-packed fragment-major buffer:
// dword index = ((kt*ntiles + nt)*32 + lane)*8 + j
__device__ __forceinline__ v16h load_bfrag(const unsigned* __restrict__ Bp,
                                           size_t kt, size_t nt, size_t ntiles, int lane) {
  const unsigned* p = Bp + (((kt * ntiles + nt) * 32) + (size_t)lane) * 8;
  Frag16 f;
  f.q[0] = *(const uint4*)(p);
  f.q[1] = *(const uint4*)(p + 4);
  return f.v;
}

// ---------------------------------------------------------------------------
// Pack f32 weights into WMMA B-fragment layout (f16).
// mode 0: W is [N,K] row-major (B[k,n] = W[n,k])  -- all plain "x @ W^T" GEMMs
// mode 1: conv_in [256,128,5]: k = k5*128 + i ; B[k,n] = W[n,i,k5]
__global__ void pack_b_kernel(const float* __restrict__ W, unsigned* __restrict__ out,
                              int K, int N, int mode) {
  size_t idx = (size_t)blockIdx.x * blockDim.x + threadIdx.x;
  size_t total = (size_t)K * N / 2;
  if (idx >= total) return;
  int j    = (int)(idx & 7);
  int lane = (int)((idx >> 3) & 31);
  size_t tile = idx >> 8;
  int ntiles = N >> 4;
  int nt = (int)(tile % ntiles);
  int kt = (int)(tile / ntiles);
  int n  = nt * 16 + (lane & 15);
  int kloc = ((lane >> 4) * 8) + ((j < 4) ? 2 * j : 16 + 2 * (j - 4));
  int k  = kt * 32 + kloc;
  float w0, w1;
  if (mode == 0) {
    w0 = W[(size_t)n * K + k];
    w1 = W[(size_t)n * K + k + 1];
  } else {
    int ka = k, kb = k + 1;
    w0 = W[((size_t)n * 128 + (ka & 127)) * 5 + (ka >> 7)];
    w1 = W[((size_t)n * 128 + (kb & 127)) * 5 + (kb >> 7)];
  }
  unsigned short h0 = __builtin_bit_cast(unsigned short, (half_t)w0);
  unsigned short h1 = __builtin_bit_cast(unsigned short, (half_t)w1);
  out[idx] = (unsigned)h0 | ((unsigned)h1 << 16);
}

// ---------------------------------------------------------------------------
// Generic WMMA GEMM: Out[b*T+t, n] = epi( sum_k A_row(b,t)[k] * B[k,n] )
// Software-pipelined: fragments for kt+1 are issued before the WMMA for kt, so
// the pre-WMMA s_wait_loadcnt can leave next-iteration loads outstanding.
// epi: 0 = raw f16, 1 = tanh, 2 = GLU: tanh(Aux[row,n]) * sigmoid(acc)
__global__ __launch_bounds__(256)
void gemm_wmma_f16(const half_t* __restrict__ A, const unsigned* __restrict__ Bp,
                   half_t* __restrict__ Out, const half_t* __restrict__ Aux,
                   int ArowsPerB, int Astride, int K, int N, int mtiles, int epi) {
  const int lane = threadIdx.x & 31;
  const int wv = blockIdx.x * (blockDim.x >> 5) + (threadIdx.x >> 5);
  const int ntiles = N >> 4;
  if (wv >= mtiles * ntiles) return;          // wave-uniform exit (EXEC stays full)
  const int mt = wv / ntiles;
  const int nt = wv - mt * ntiles;
  const int mrow = lane & 15;
  const int r = mt * 16 + mrow;
  const int b = r / T_;
  const int t = r - b * T_;
  const half_t* rowp = A + ((size_t)b * ArowsPerB + t) * (size_t)Astride;
  const int kb = (lane >> 4) * 8;
  const int ktiles = K >> 5;
  v8f acc = {0.f, 0.f, 0.f, 0.f, 0.f, 0.f, 0.f, 0.f};
  v16h a  = load_a(rowp, kb);
  v16h bf = load_bfrag(Bp, 0, nt, ntiles, lane);
  for (int kt = 0; kt + 1 < ktiles; ++kt) {
    v16h a2 = load_a(rowp, (kt + 1) * 32 + kb);
    v16h b2 = load_bfrag(Bp, (size_t)(kt + 1), nt, ntiles, lane);
    __builtin_prefetch(Bp + (((size_t)(kt + 2) * ntiles + nt) * 32 + lane) * 8, 0, 1);
    acc = __builtin_amdgcn_wmma_f32_16x16x32_f16(false, a, false, bf, (short)0, acc,
                                                 false, false);
    a = a2; bf = b2;
  }
  acc = __builtin_amdgcn_wmma_f32_16x16x32_f16(false, a, false, bf, (short)0, acc,
                                               false, false);
  // C layout: lane n = L%16 ; VGPR v -> m = v + 8*(L/16)
  const int row_l = 8 * (lane >> 4);
  const int col = nt * 16 + (lane & 15);
#pragma unroll
  for (int v = 0; v < 8; ++v) {
    size_t rr = (size_t)(mt * 16 + row_l + v);
    size_t oidx = rr * (size_t)N + col;
    float x = acc[v];
    float o;
    if (epi == 0)      o = x;
    else if (epi == 1) o = tanhf(x);
    else               o = tanhf((float)Aux[oidx]) * sigmoidf_(x);
    Out[oidx] = (half_t)o;
  }
}

// ---------------------------------------------------------------------------
// Persistent GRU: 2 blocks x 16 batch rows; 8 waves; h in LDS (f16) + regs (f32).
// Wave w owns column groups {16w, 16(w+8)} and the matching r/z/n tiles so the
// gate math is fully register-local. 48 WMMAs / step / wave; W_hh streamed.
// gi gate inputs are double-buffered in LDS via global_load_async_to_lds_b128:
// each wave async-prefetches (one step ahead) exactly the 6 column-blocks x 16
// rows it will read itself -> per-wave s_wait_asynccnt, no extra barriers.
__global__ __launch_bounds__(256)
void gru_kernel(const half_t* __restrict__ gi, const unsigned* __restrict__ Bhh,
                const half_t* __restrict__ h0, half_t* __restrict__ gout) {
  __shared__ __align__(16) half_t hsh[16 * 256];
  __shared__ __align__(16) half_t gish[2][16 * 768];
  const int lane = threadIdx.x & 31;
  const int w = threadIdx.x >> 5;
  const int bbase = blockIdx.x * 16;
  for (int i = threadIdx.x; i < 16 * 256; i += 256)
    hsh[i] = h0[(size_t)(bbase + (i >> 8)) * 256 + (i & 255)];
  __syncthreads();
  const int colg[2] = {16 * w, 16 * (w + 8)};
  const int row_l = 8 * (lane >> 4);
  const int nsub = lane & 15;
  v8f hreg[2];
#pragma unroll
  for (int g = 0; g < 2; ++g)
#pragma unroll
    for (int v = 0; v < 8; ++v)
      hreg[g][v] = (float)hsh[(row_l + v) * 256 + colg[g] + nsub];

  const unsigned gish_base = (unsigned)(uintptr_t)&gish[0][0];
  // Async prefetch of gi[ bbase..bbase+15 , wave-owned cols ] for step t -> buf.
  auto issue_gi = [&](int t, int buf) {
#pragma unroll
    for (int i = 0; i < 6; ++i) {
      int cid = lane + 32 * i;             // 0..191 -> (row 0..15, sub 0..11)
      int row = cid / 12;
      int sub = cid - row * 12;
      int cb  = sub >> 1;                  // 0..5 : {g0,g0+256,g0+512,g1,...}
      int colstart = ((cb < 3) ? (colg[0] + cb * 256)
                               : (colg[1] + (cb - 3) * 256)) + (sub & 1) * 8;
      const half_t* src = gi + ((size_t)(bbase + row) * T_ + t) * 768 + colstart;
      unsigned lds = gish_base +
                     (unsigned)((buf * (16 * 768) + row * 768 + colstart) * 2);
      asm volatile("global_load_async_to_lds_b128 %0, %1, off"
                   :: "v"(lds), "v"(src) : "memory");
    }
  };
  issue_gi(0, 0);

  const int kb = (lane >> 4) * 8;
  const half_t* rowp = &hsh[(lane & 15) * 256];
  const v8f vz = {0.f, 0.f, 0.f, 0.f, 0.f, 0.f, 0.f, 0.f};
  for (int t = 0; t < T_; ++t) {
    bool more = (t + 1 < T_);
    if (more) issue_gi(t + 1, (t + 1) & 1);
    v16h afr[8];
#pragma unroll
    for (int kt = 0; kt < 8; ++kt) afr[kt] = load_a(rowp, kt * 32 + kb);
    v8f acc[2][3];
#pragma unroll
    for (int g = 0; g < 2; ++g) {
#pragma unroll
      for (int p = 0; p < 3; ++p) acc[g][p] = vz;
    }
#pragma unroll
    for (int g = 0; g < 2; ++g) {
      int ntbase = colg[g] >> 4;
#pragma unroll
      for (int p = 0; p < 3; ++p) {
        size_t nt = (size_t)(ntbase + p * 16);       // r | z | n tile columns
#pragma unroll
        for (int kt = 0; kt < 8; ++kt) {
          v16h bf = load_bfrag(Bhh, (size_t)kt, nt, 48, lane);
          acc[g][p] = __builtin_amdgcn_wmma_f32_16x16x32_f16(false, afr[kt], false, bf,
                                                             (short)0, acc[g][p],
                                                             false, false);
        }
      }
    }
    __syncthreads();                // all lanes done reading old h from LDS
    // wait for this step's gi (allow the 6 loads for t+1 to stay in flight)
    if (more) asm volatile("s_wait_asynccnt 0x6" ::: "memory");
    else      asm volatile("s_wait_asynccnt 0x0" ::: "memory");
    const half_t* gb = &gish[t & 1][0];
#pragma unroll
    for (int g = 0; g < 2; ++g) {
      int col = colg[g] + nsub;
#pragma unroll
      for (int v = 0; v < 8; ++v) {
        int m = row_l + v;
        int bg = bbase + m;
        float gr = (float)gb[m * 768 + col];
        float gz = (float)gb[m * 768 + col + 256];
        float gn = (float)gb[m * 768 + col + 512];
        float rv = sigmoidf_(gr + acc[g][0][v]);
        float zv = sigmoidf_(gz + acc[g][1][v]);
        float nv = tanhf(gn + rv * acc[g][2][v]);
        float hv = (1.0f - zv) * nv + zv * hreg[g][v];
        hreg[g][v] = hv;
        half_t hh = (half_t)hv;
        hsh[m * 256 + col] = hh;
        gout[((size_t)bg * T_ + t) * 256 + col] = hh;
      }
    }
    __syncthreads();                // new h visible before next step's loads
  }
}

// ---------------------------------------------------------------------------
// Front-end / small kernels
__global__ void phase0_kernel(const float* __restrict__ pp, float* __restrict__ ph0) {
  int b = blockIdx.x * blockDim.x + threadIdx.x;
  if (b >= B_) return;
  float acc = 0.0f;
  for (int s = 0; s < S_; ++s) {
    ph0[b * S_ + s] = acc;
    acc += 160.0f * (2.0f * PI_ / pp[b * S_ + s]);
  }
}

__global__ void feat_zero_kernel(half_t* feat) {
  int idx = blockIdx.x * blockDim.x + threadIdx.x;
  if (idx >= B_ * 4 * 128) return;
  int c = idx & 127; int rI = (idx >> 7) & 3; int b = idx >> 9;
  int row = (rI < 3) ? rI : 2003;          // left pad rows 0..2, right pad row 2003
  feat[((size_t)b * 2004 + row) * 128 + c] = (half_t)0.0f;
}

__global__ void phase_fill_kernel(const float* __restrict__ pp, const float* __restrict__ ph0,
                                  half_t* __restrict__ feat) {
  size_t idx = (size_t)blockIdx.x * blockDim.x + threadIdx.x;
  if (idx >= (size_t)B_ * S_ * 160) return;
  int i = (int)(idx % 160); size_t sb = idx / 160;
  int s = (int)(sb % S_); int b = (int)(sb / S_);
  float f = 2.0f * PI_ / pp[b * S_ + s];
  float arg = f * (float)(i + 1) + ph0[b * S_ + s];
  float sv, cv;
  sincosf(arg, &sv, &cv);
  int t = s * 5 + (i >> 5); int c = i & 31;
  half_t* base = feat + ((size_t)b * 2004 + 3 + t) * 128;
  base[c] = (half_t)sv;
  base[32 + c] = (half_t)cv;
}

__global__ void env_kernel(const float* __restrict__ bfcc, const float* __restrict__ wup,
                           half_t* __restrict__ feat) {
  size_t idx = (size_t)blockIdx.x * blockDim.x + threadIdx.x;
  if (idx >= (size_t)B_ * T_ * 64) return;
  int c = (int)(idx & 63); size_t tb = idx >> 6;
  int t = (int)(tb % T_); int b = (int)(tb / T_);
  int s = t / 5, k = t - s * 5;
  float acc = 0.0f;
  const float* x = bfcc + ((size_t)b * S_ + s) * 19;
#pragma unroll
  for (int i = 0; i < 19; ++i) acc += x[i] * wup[(i * 64 + c) * 5 + k];
  feat[((size_t)b * 2004 + 3 + t) * 128 + 64 + c] = (half_t)tanhf(acc);
}

__global__ void h0_kernel(const float* __restrict__ x0, const float* __restrict__ w,
                          half_t* __restrict__ h0) {
  int idx = blockIdx.x * blockDim.x + threadIdx.x;
  if (idx >= B_ * 256) return;
  int n = idx & 255, b = idx >> 8;
  float acc = 0.0f;
  const float* x = x0 + b * 320;
  const float* wr = w + (size_t)n * 320;
  for (int k = 0; k < 320; ++k) acc += x[k] * wr[k];
  h0[idx] = (half_t)tanhf(acc);
}

// Build padded frame buffer P[b, T+2, fl]: rows 0..1 = tanh(x0 @ wc^T), rows 2.. = in.
__global__ void fwc_pad_kernel(const float* __restrict__ x0, const float* __restrict__ wc,
                               const half_t* __restrict__ in, half_t* __restrict__ P, int fl) {
  size_t idx = (size_t)blockIdx.x * blockDim.x + threadIdx.x;
  size_t total = (size_t)B_ * (T_ + 2) * fl;
  if (idx >= total) return;
  int c = (int)(idx % fl); size_t rb = idx / fl;
  int row = (int)(rb % (T_ + 2)); int b = (int)(rb / (T_ + 2));
  half_t val;
  if (row < 2) {
    float acc = 0.0f;
    const float* x = x0 + b * 320;
    const float* wr = wc + (size_t)(row * fl + c) * 320;
    for (int k = 0; k < 320; ++k) acc += x[k] * wr[k];
    val = (half_t)tanhf(acc);
  } else {
    val = in[((size_t)b * T_ + (row - 2)) * (size_t)fl + c];
  }
  P[idx] = val;
}

__global__ void out_gain_kernel(const half_t* __restrict__ act, const float* __restrict__ bfcc,
                                float* __restrict__ out) {
  size_t idx = (size_t)blockIdx.x * blockDim.x + threadIdx.x;
  if (idx >= (size_t)B_ * S_ * 160) return;
  int j = (int)(idx % (S_ * 160)); int b = (int)(idx / (S_ * 160));
  int s = j / 160; int t = j >> 5; int c = j & 31;
  float wav = (float)act[((size_t)b * T_ + t) * 32 + c];
  float e = 0.5f * bfcc[((size_t)b * S_ + s) * 19] / sqrtf(18.0f);
  out[idx] = wav * exp10f(e);
}

// ---------------------------------------------------------------------------
static inline int gblk(size_t total, int bs) { return (int)((total + bs - 1) / bs); }

static void run_gemm(hipStream_t st, const half_t* A, const unsigned* Bp, half_t* Out,
                     const half_t* Aux, int ArowsPerB, int Astride, int K, int N, int epi) {
  int mtiles = (B_ * T_) / 16;                          // 4000
  long waves = (long)mtiles * (N >> 4);
  int blocks = (int)((waves + 7) / 8);                  // 8 waves (256 thr) per block
  gemm_wmma_f16<<<blocks, 256, 0, st>>>(A, Bp, Out, Aux, ArowsPerB, Astride, K, N, mtiles, epi);
}

static void run_pack(hipStream_t st, const float* W, unsigned* out, int K, int N, int mode) {
  size_t total = (size_t)K * N / 2;
  pack_b_kernel<<<gblk(total, 256), 256, 0, st>>>(W, out, K, N, mode);
}

extern "C" void kernel_launch(void* const* d_in, const int* in_sizes, int n_in,
                              void* d_out, int out_size, void* d_ws, size_t ws_size,
                              hipStream_t stream) {
  (void)in_sizes; (void)n_in; (void)out_size; (void)ws_size;
  const float* pp     = (const float*)d_in[0];
  const float* bfcc   = (const float*)d_in[1];
  const float* x0     = (const float*)d_in[2];
  const float* w_up   = (const float*)d_in[3];
  const float* w_conv = (const float*)d_in[4];
  const float* w_glu  = (const float*)d_in[5];
  const float* w_cont = (const float*)d_in[6];
  const float* w_ih   = (const float*)d_in[7];
  const float* w_hh   = (const float*)d_in[8];
  const float* w_rglu = (const float*)d_in[9];
  const float* fwc_cont[7]; const float* fwc_fc[7]; const float* fwc_gate[6];
  {
    int idx = 10;
    for (int i = 0; i < 7; ++i) {
      fwc_cont[i] = (const float*)d_in[idx++];
      fwc_fc[i]   = (const float*)d_in[idx++];
      if (i < 6) fwc_gate[i] = (const float*)d_in[idx++];
    }
  }

  const size_t MiB = 1ull << 20;
  char* ws = (char*)d_ws;
  half_t* feat   = (half_t*)(ws + 0);           // [B, 2004, 128] f16 im2col (16.4 MB)
  half_t* hconv  = (half_t*)(ws + 16 * MiB);    // conv linear out; later fwc actA
  half_t* wavlat = (half_t*)(ws + 48 * MiB);    // wav_latent; later fwc padded buf
  half_t* gi     = (half_t*)(ws + 80 * MiB);    // [B*T, 768] f16 (98 MB)
  half_t* lin    = (half_t*)(ws + 80 * MiB);    // fwc linear (reuses gi after GRU)
  half_t* actB   = (half_t*)(ws + 112 * MiB);   // fwc act ping-pong (inside gi region)
  half_t* gruo   = (half_t*)(ws + 176 * MiB);   // GRU output
  half_t* rnno   = (half_t*)(ws + 208 * MiB);   // rnn GLU output
  half_t* padP   = wavlat;
  half_t* actA   = hconv;
  half_t* h0b    = (half_t*)(ws + 240 * MiB);
  float*  ph0    = (float*)(ws + 240 * MiB + 64 * 1024);
  unsigned* packs = (unsigned*)(ws + 241 * MiB);

  size_t poff = 0;
  auto palloc = [&](int K, int N) { unsigned* p = packs + poff; poff += (size_t)K * N / 2; return p; };
  unsigned* p_conv = palloc(640, 256);
  unsigned* p_glu  = palloc(256, 256);
  unsigned* p_ih   = palloc(256, 768);
  unsigned* p_hh   = palloc(256, 768);
  unsigned* p_rglu = palloc(256, 256);
  const int fls[7]  = {256, 256, 128, 128, 64, 64, 32};
  const int outs[7] = {256, 128, 128, 64, 64, 32, 32};
  unsigned* p_fc[7]; unsigned* p_gt[7];
  for (int i = 0; i < 7; ++i) {
    p_fc[i] = palloc(3 * fls[i], outs[i]);
    p_gt[i] = (i < 6) ? palloc(outs[i], outs[i]) : nullptr;
  }

  // ---- weight packing (f32 -> WMMA f16 B fragments) ----
  run_pack(stream, w_conv, p_conv, 640, 256, 1);
  run_pack(stream, w_glu,  p_glu,  256, 256, 0);
  run_pack(stream, w_ih,   p_ih,   256, 768, 0);
  run_pack(stream, w_hh,   p_hh,   256, 768, 0);
  run_pack(stream, w_rglu, p_rglu, 256, 256, 0);
  for (int i = 0; i < 7; ++i) {
    run_pack(stream, fwc_fc[i], p_fc[i], 3 * fls[i], outs[i], 0);
    if (i < 6) run_pack(stream, fwc_gate[i], p_gt[i], outs[i], outs[i], 0);
  }

  // ---- front-end ----
  feat_zero_kernel<<<gblk((size_t)B_ * 4 * 128, 256), 256, 0, stream>>>(feat);
  phase0_kernel<<<1, 32, 0, stream>>>(pp, ph0);
  phase_fill_kernel<<<gblk((size_t)B_ * S_ * 160, 256), 256, 0, stream>>>(pp, ph0, feat);
  env_kernel<<<gblk((size_t)B_ * T_ * 64, 256), 256, 0, stream>>>(bfcc, w_up, feat);
  h0_kernel<<<gblk((size_t)B_ * 256, 256), 256, 0, stream>>>(x0, w_cont, h0b);

  // ---- conv lookahead (im2col GEMM K=640) + GLU ----
  run_gemm(stream, feat,  p_conv, hconv,  nullptr, 2004, 128, 640, 256, 0);
  run_gemm(stream, hconv, p_glu,  wavlat, hconv,   T_,   256, 256, 256, 2);

  // ---- GRU: input projection then persistent recurrence ----
  run_gemm(stream, wavlat, p_ih, gi, nullptr, T_, 256, 256, 768, 0);
  gru_kernel<<<2, 256, 0, stream>>>(gi, p_hh, h0b, gruo);
  run_gemm(stream, gruo, p_rglu, rnno, gruo, T_, 256, 256, 256, 2);

  // ---- framewise conv stack ----
  const half_t* cur = rnno;
  half_t* acts2[2] = {actA, actB};
  for (int i = 0; i < 7; ++i) {
    int fl = fls[i], outc = outs[i];
    fwc_pad_kernel<<<gblk((size_t)B_ * (T_ + 2) * fl, 256), 256, 0, stream>>>(
        x0, fwc_cont[i], cur, padP, fl);
    if (i < 6) {
      run_gemm(stream, padP, p_fc[i], lin, nullptr, T_ + 2, fl, 3 * fl, outc, 0);
      run_gemm(stream, lin, p_gt[i], acts2[i & 1], lin, T_, outc, outc, outc, 2);
    } else {
      run_gemm(stream, padP, p_fc[i], acts2[i & 1], nullptr, T_ + 2, fl, 3 * fl, outc, 1);
    }
    cur = acts2[i & 1];
  }

  // ---- gain + output ----
  out_gain_kernel<<<gblk((size_t)B_ * S_ * 160, 256), 256, 0, stream>>>(
      cur, bfcc, (float*)d_out);
}
